// A2aSparseStackedMlp_65833258713875
// MI455X (gfx1250) — matrix operand order repaired and checked by reading, not compiled, not run on hardware
//
#include <hip/hip_runtime.h>
#include <hip/hip_bf16.h>

#define K_ALPHA 1.702f
#define K_LIMIT 7.0f

// Problem sizes (B,S,H,E,I) = (2,1024,768,8,768)
constexpr int T  = 2048;        // B*S tokens
constexpr int Hh = 768;
constexpr int Ee = 8;
constexpr int Ii = 768;
constexpr int Ff = 1536;        // 2*I

constexpr int MT = 32;          // token tile (M)
constexpr int LP = 776;         // padded LDS row in bf16 elems (1552B: bank-spread, 16B-aligned)

typedef __attribute__((ext_vector_type(16))) __bf16 bf16x16;
typedef __attribute__((ext_vector_type(8)))  float  f32x8;

union BV { bf16x16 v; uint4 u[2]; };

// ---------------------------------------------------------------------------
// A operand (16x32 bf16, row-major in LDS): lane m<16 holds M=m, K in
// {k0..k0+7, k0+16..k0+23}; lane m+16 holds M=m, K in {k0+8..15, k0+24..31}.
__device__ inline bf16x16 load_a(const __hip_bfloat16* row_ptr, int k0, int sel) {
  const __hip_bfloat16* p = row_ptr + k0 + (sel ? 8 : 0);
  BV r;
  r.u[0] = *(const uint4*)(p);
  r.u[1] = *(const uint4*)(p + 16);
  return r.v;
}

// B operand (32x16 bf16) from a column-major (n-major, K contiguous) weight:
// lane n<16 holds N=n, K in {k0..k0+7, k0+16..k0+23}; lane n+16 holds N=n, K in {k0+8..15, k0+24..31}.
__device__ inline bf16x16 load_b(const __hip_bfloat16* col_ptr, int k0, int sel) {
  const __hip_bfloat16* p = col_ptr + k0 + (sel ? 8 : 0);
  BV r;
  r.u[0] = *(const uint4*)(p);
  r.u[1] = *(const uint4*)(p + 16);
  return r.v;
}

__device__ inline f32x8 wmma_bf16(bf16x16 a, bf16x16 b, f32x8 c) {
  return __builtin_amdgcn_wmma_f32_16x16x32_bf16(false, a, false, b,
                                                 (short)0, c, false, false);
}

// ---------------------------------------------------------------------------
// x (activations) f32 -> bf16, coalesced both sides.
__global__ void cvt_x_kernel(const float* __restrict__ src,
                             __hip_bfloat16* __restrict__ dst, int n) {
  int i = blockIdx.x * blockDim.x + threadIdx.x;
  if (i < n) dst[i] = __float2bfloat16(src[i]);
}

// Tiled transpose + convert: per e, src is [R][C] f32 row-major,
// dst is [C][R] bf16 (so WMMA-B K-dim is contiguous). 32x32 tiles via LDS,
// coalesced reads and writes. R, C multiples of 32.
__global__ void transpose_cvt_kernel(const float* __restrict__ src,
                                     __hip_bfloat16* __restrict__ dst,
                                     int R, int C) {
  __shared__ float tile[32][33];
  const int e = blockIdx.z;
  const float* s = src + (size_t)e * R * C;
  __hip_bfloat16* d = dst + (size_t)e * R * C;
  const int c0 = blockIdx.x * 32;
  const int r0 = blockIdx.y * 32;
  const int tx = threadIdx.x;      // 0..31
  const int ty = threadIdx.y;      // 0..7
#pragma unroll
  for (int j = 0; j < 4; ++j)
    tile[ty + j * 8][tx] = s[(size_t)(r0 + ty + j * 8) * C + (c0 + tx)];
  __syncthreads();
#pragma unroll
  for (int j = 0; j < 4; ++j)
    d[(size_t)(c0 + ty + j * 8) * R + (r0 + tx)] =
        __float2bfloat16(tile[tx][ty + j * 8]);
}

// ---------------------------------------------------------------------------
// Router: one wave32 per token. Dot(x, Wr) per expert, top-2, softmax,
// append (token, weight) to per-expert compacted lists.
__global__ void router_kernel(const float* __restrict__ x,
                              const float* __restrict__ Wr,
                              const float* __restrict__ br,
                              int* __restrict__ counts,
                              int* __restrict__ tok_list,
                              float* __restrict__ wt_list) {
  int wave = threadIdx.x >> 5;
  int lane = threadIdx.x & 31;
  int t = blockIdx.x * (blockDim.x >> 5) + wave;
  if (t >= T) return;

  float acc[Ee];
#pragma unroll
  for (int e = 0; e < Ee; ++e) acc[e] = 0.f;

  const float* xrow = x + (size_t)t * Hh;
  for (int h = lane; h < Hh; h += 32) {
    float xv = xrow[h];
    const float* wr = Wr + (size_t)h * Ee;
#pragma unroll
    for (int e = 0; e < Ee; ++e) acc[e] = fmaf(xv, wr[e], acc[e]);
  }
#pragma unroll
  for (int off = 16; off > 0; off >>= 1) {
#pragma unroll
    for (int e = 0; e < Ee; ++e) acc[e] += __shfl_xor(acc[e], off, 32);
  }

  if (lane == 0) {
    float lg[Ee];
#pragma unroll
    for (int e = 0; e < Ee; ++e) lg[e] = acc[e] + br[e];
    // top-2 (first index wins ties, matching lax.top_k)
    int e0 = 0; float v0 = lg[0];
#pragma unroll
    for (int e = 1; e < Ee; ++e) { if (lg[e] > v0) { v0 = lg[e]; e0 = e; } }
    int e1 = -1; float v1 = -3.4e38f;
#pragma unroll
    for (int e = 0; e < Ee; ++e) {
      if (e != e0 && lg[e] > v1) { v1 = lg[e]; e1 = e; }
    }
    float m  = fmaxf(v0, v1);
    float a0 = __expf(v0 - m), a1 = __expf(v1 - m);
    float inv = 1.f / (a0 + a1);
    float w0 = a0 * inv, w1 = a1 * inv;

    int p0 = atomicAdd(&counts[e0], 1);
    tok_list[(size_t)e0 * T + p0] = t;
    wt_list [(size_t)e0 * T + p0] = w0;
    int p1 = atomicAdd(&counts[e1], 1);
    tok_list[(size_t)e1 * T + p1] = t;
    wt_list [(size_t)e1 * T + p1] = w1;
  }
}

// ---------------------------------------------------------------------------
// Fused expert MLP over a 32-token tile of one expert's token list.
// block = 256 (8 waves). Wave w owns gate cols [w*96, w*96+96), the matching
// up cols (+768), and out cols [w*96, w*96+96). Two 16-row M-tiles per wave
// reuse every B fragment twice (halves L2 weight traffic).
__global__ void __launch_bounds__(256)
moe_kernel(const __hip_bfloat16* __restrict__ xb,
           const __hip_bfloat16* __restrict__ wgu_t,   // [e][f][h]
           const __hip_bfloat16* __restrict__ wd_t,    // [e][h][i]
           const float* __restrict__ bgu,              // [e][2I]
           const float* __restrict__ bd,               // [e][H]
           const int*   __restrict__ counts,
           const int*   __restrict__ tok_list,
           const float* __restrict__ wt_list,
           float* __restrict__ out) {
  const int e    = blockIdx.y;
  const int cnt  = counts[e];
  const int base = blockIdx.x * MT;
  if (base >= cnt) return;

  __shared__ __hip_bfloat16 xa [MT][LP];   // 48.5 KB (row-padded, bank-spread)
  __shared__ __hip_bfloat16 act[MT][LP];   // 48.5 KB
  __shared__ int   s_tok[MT];
  __shared__ float s_w[MT];

  const int wave = threadIdx.x >> 5;
  const int lane = threadIdx.x & 31;
  const int nl   = lane & 15;
  const int sel  = (lane >> 4) & 1;

  if (threadIdx.x < MT) {
    int idx = base + threadIdx.x;
    if (idx < cnt) {
      s_tok[threadIdx.x] = tok_list[(size_t)e * T + idx];
      s_w[threadIdx.x]   = wt_list [(size_t)e * T + idx];
    } else {
      s_tok[threadIdx.x] = 0;     // valid row, weight 0 => contributes nothing
      s_w[threadIdx.x]   = 0.f;
    }
  }
  __syncthreads();

  // Gather x tile: 4 rows per wave, 128-bit global loads + LDS writes.
#pragma unroll
  for (int rr = 0; rr < 4; ++rr) {
    int row = wave + rr * 8;
    const uint4* src = (const uint4*)(xb + (size_t)s_tok[row] * Hh);
    uint4* dstp = (uint4*)(&xa[row][0]);
    for (int i = lane; i < Hh / 8; i += 32) dstp[i] = src[i];
  }
  __syncthreads();

  const __hip_bfloat16* wgu_e = wgu_t + (size_t)e * Ff * Hh;
  const __hip_bfloat16* wd_e  = wd_t  + (size_t)e * Hh * Ii;
  const float* bgu_e = bgu + (size_t)e * Ff;
  const float* bd_e  = bd  + (size_t)e * Hh;

  // ---------------- GEMM1 (x @ Wgu) + clamp + gated activation -------------
  for (int g = 0; g < 3; ++g) {
    const int gate_base = wave * 96 + g * 32;       // [0,768)
    const int up_base   = gate_base + Ii;           // [768,1536)
    f32x8 c[8] = {};                                // [mtile*4 + ntile]
    for (int kk = 0; kk < 24; ++kk) {
      const int k0 = kk * 32;
      bf16x16 a0 = load_a(&xa[nl][0],      k0, sel);
      bf16x16 a1 = load_a(&xa[nl + 16][0], k0, sel);
      bf16x16 b0 = load_b(wgu_e + (size_t)(gate_base +      nl) * Hh, k0, sel);
      bf16x16 b1 = load_b(wgu_e + (size_t)(gate_base + 16 + nl) * Hh, k0, sel);
      bf16x16 b2 = load_b(wgu_e + (size_t)(up_base   +      nl) * Hh, k0, sel);
      bf16x16 b3 = load_b(wgu_e + (size_t)(up_base   + 16 + nl) * Hh, k0, sel);
      c[0] = wmma_bf16(a0, b0, c[0]);
      c[1] = wmma_bf16(a0, b1, c[1]);
      c[2] = wmma_bf16(a0, b2, c[2]);
      c[3] = wmma_bf16(a0, b3, c[3]);
      c[4] = wmma_bf16(a1, b0, c[4]);
      c[5] = wmma_bf16(a1, b1, c[5]);
      c[6] = wmma_bf16(a1, b2, c[6]);
      c[7] = wmma_bf16(a1, b3, c[7]);
    }
#pragma unroll
    for (int mt = 0; mt < 2; ++mt) {
#pragma unroll
      for (int tI = 0; tI < 2; ++tI) {
        const int gcol = gate_base + tI * 16 + nl;
        const float bg = bgu_e[gcol];
        const float bu = bgu_e[gcol + Ii];
#pragma unroll
        for (int r = 0; r < 8; ++r) {
          const int m = mt * 16 + r + (sel ? 8 : 0);
          float gate = fminf(c[mt * 4 + tI][r] + bg, K_LIMIT);
          float up   = fmaxf(fminf(c[mt * 4 + tI + 2][r] + bu, K_LIMIT), -K_LIMIT);
          float glu  = gate / (1.f + __expf(-gate * K_ALPHA));   // x*sigmoid(ax)
          act[m][gcol] = __float2bfloat16((up + 1.f) * glu);
        }
      }
    }
  }
  __syncthreads();

  // ---------------- GEMM2 (act @ Wd) + bias + weighted scatter -------------
  for (int g2 = 0; g2 < 2; ++g2) {
    const int nb = wave * 96 + g2 * 48;
    f32x8 c[6] = {};                                // [mtile*3 + ntile]
    for (int kk = 0; kk < 24; ++kk) {
      const int k0 = kk * 32;
      bf16x16 a0 = load_a(&act[nl][0],      k0, sel);
      bf16x16 a1 = load_a(&act[nl + 16][0], k0, sel);
      bf16x16 b0 = load_b(wd_e + (size_t)(nb +      nl) * Ii, k0, sel);
      bf16x16 b1 = load_b(wd_e + (size_t)(nb + 16 + nl) * Ii, k0, sel);
      bf16x16 b2 = load_b(wd_e + (size_t)(nb + 32 + nl) * Ii, k0, sel);
      c[0] = wmma_bf16(a0, b0, c[0]);
      c[1] = wmma_bf16(a0, b1, c[1]);
      c[2] = wmma_bf16(a0, b2, c[2]);
      c[3] = wmma_bf16(a1, b0, c[3]);
      c[4] = wmma_bf16(a1, b1, c[4]);
      c[5] = wmma_bf16(a1, b2, c[5]);
    }
#pragma unroll
    for (int mt = 0; mt < 2; ++mt) {
#pragma unroll
      for (int j = 0; j < 3; ++j) {
        const int col = nb + j * 16 + nl;
        const float bdv = bd_e[col];
#pragma unroll
        for (int r = 0; r < 8; ++r) {
          const int m = mt * 16 + r + (sel ? 8 : 0);
          const float v = (c[mt * 3 + j][r] + bdv) * s_w[m];
          atomicAdd(&out[(size_t)s_tok[m] * Hh + col], v);
        }
      }
    }
  }
}

// ---------------------------------------------------------------------------
extern "C" void kernel_launch(void* const* d_in, const int* in_sizes, int n_in,
                              void* d_out, int out_size, void* d_ws, size_t ws_size,
                              hipStream_t stream) {
  const float* x   = (const float*)d_in[0];   // (B,S,H)
  const float* Wr  = (const float*)d_in[1];   // (H,E)
  const float* br  = (const float*)d_in[2];   // (E)
  const float* Wgu = (const float*)d_in[3];   // (E,H,2I)
  const float* bgu = (const float*)d_in[4];   // (E,2I)
  const float* Wd  = (const float*)d_in[5];   // (E,I,H)
  const float* bd  = (const float*)d_in[6];   // (E,H)
  float* out = (float*)d_out;

  // Workspace carve-up (~31.6 MB total)
  char* ws = (char*)d_ws;
  int*   counts   = (int*)ws;                           // 64 B (8 used)
  int*   tok_list = (int*)(ws + 64);                    // 8*2048*4 = 64 KB
  float* wt_list  = (float*)(ws + 64 + 65536);          // 64 KB
  __hip_bfloat16* xb    = (__hip_bfloat16*)(ws + 64 + 2 * 65536);
  __hip_bfloat16* wgu_t = xb + (size_t)T * Hh;          // 3 MB after xb
  __hip_bfloat16* wd_t  = wgu_t + (size_t)Ee * Ff * Hh; // 18.9 MB after wgu_t
  (void)in_sizes; (void)n_in; (void)ws_size;

  hipMemsetAsync(counts, 0, 64, stream);
  hipMemsetAsync(out, 0, (size_t)out_size * sizeof(float), stream);

  { int n = T * Hh;
    cvt_x_kernel<<<(n + 255) / 256, 256, 0, stream>>>(x, xb, n); }

  // Wgu: per-e [H][F] -> [F][H];  Wd: per-e [I][H] -> [H][I]
  transpose_cvt_kernel<<<dim3(Ff / 32, Hh / 32, Ee), dim3(32, 8), 0, stream>>>(
      Wgu, wgu_t, Hh, Ff);
  transpose_cvt_kernel<<<dim3(Hh / 32, Ii / 32, Ee), dim3(32, 8), 0, stream>>>(
      Wd, wd_t, Ii, Hh);

  router_kernel<<<T / 8, 256, 0, stream>>>(x, Wr, br, counts, tok_list, wt_list);

  dim3 grid(T / MT, Ee);   // 64 token tiles x 8 experts (early-exit past count)
  moe_kernel<<<grid, 256, 0, stream>>>(xb, wgu_t, wd_t, bgu, bd,
                                       counts, tok_list, wt_list, out);
}
